// NonCausalSelfAttentionTransformerBlock_46626164965960
// MI455X (gfx1250) — matrix operand order repaired and checked by reading, not compiled, not run on hardware
//
#include <hip/hip_runtime.h>
#include <hip/hip_bf16.h>

// Shapes from the reference
#define BB 2
#define SS 2048
#define DD 768
#define HH 12
#define DHE 64
#define FF 3072

typedef __attribute__((ext_vector_type(16))) _Float16 v16h;
typedef __attribute__((ext_vector_type(8)))  _Float16 v8h;
typedef __attribute__((ext_vector_type(8)))  float    v8f;
typedef __attribute__((ext_vector_type(4)))  int      v4i;

union Frag16 { v16h v; v8h h[2]; };

#if __has_builtin(__builtin_amdgcn_global_load_async_to_lds_b128) && \
    __has_builtin(__builtin_amdgcn_s_wait_asynccnt)
#define GEMM_ASYNC 1
typedef __attribute__((address_space(1))) void as1_void;
typedef __attribute__((address_space(3))) void as3_void;
typedef __attribute__((address_space(1))) v4i  as1_v4i;
typedef __attribute__((address_space(3))) v4i  as3_v4i;
#else
#define GEMM_ASYNC 0
#endif

// ---------------------------------------------------------------------------
// Tiled WMMA GEMM:  C[M,N] = A[M,K] * W[N,K]^T  (+bias[n]) (+Res[m,n])
//
// 8 waves arranged WM x WN; each wave register-blocks WMT x WNT 16x16 tiles
// (WMT A-fragments x WNT B-fragments -> WMT*WNT WMMAs per K-step, issued
// back-to-back after ALL fragments are loaded).
// Block tile = (WM*16*WMT) x (WN*16*WNT).  K stepped by 32 through LDS with
// double buffering.  Global->LDS movement uses the gfx1250 async-DMA path
// (GLOBAL_LOAD_ASYNC_TO_LDS_B128 + ASYNCcnt) when available: no VGPR staging,
// no ds_store, no spills; DMA of slab s+1 overlaps the WMMA burst of slab s.
// LDS rows padded to 40 halfs -> conflict-free ds_load_b128 fragments.
// ---------------------------------------------------------------------------
template<int WM, int WN, int WMT, int WNT, bool HAS_BIAS, bool HAS_RES>
__global__ __launch_bounds__(256)
void gemm_wmma(const _Float16* __restrict__ A, int lda,
               const _Float16* __restrict__ W, int ldb,
               float* __restrict__ C, int ldc,
               const float* __restrict__ bias,
               const float* __restrict__ Res, int ldres,
               int K)
{
    constexpr int AM  = WM * 16 * WMT;     // block rows
    constexpr int BN  = WN * 16 * WNT;     // block cols
    constexpr int LDT = 40;                // padded LDS row stride (halfs)
    constexpr int ACH = AM / 64;           // 16B chunks/thread for A tile
    constexpr int BCH = BN / 64;           // 16B chunks/thread for B tile

    __shared__ __align__(16) _Float16 As[2][AM * LDT];
    __shared__ __align__(16) _Float16 Bs[2][BN * LDT];

    const int t    = threadIdx.x;
    const int lane = t & 31;
    const int wave = t >> 5;
    const int wm   = wave / WN;
    const int wn   = wave % WN;
    const int hf   = lane >> 4;            // lane half (0/1)
    const int lr   = lane & 15;

    const int rowBlock = blockIdx.y * AM;
    const int colBlock = blockIdx.x * BN;

    const _Float16* Ag = A + (size_t)rowBlock * lda;
    const _Float16* Bg = W + (size_t)colBlock * ldb;

#if GEMM_ASYNC
    // Issue one K-slab as per-lane global->LDS async DMA (b128 per lane).
    auto aissue = [&](int k0, int buf) {
#pragma unroll
        for (int i = 0; i < ACH; ++i) {
            const int c = t + i * 256;
            const _Float16* p = Ag + (size_t)(c >> 2) * lda + (c & 3) * 8 + k0;
            _Float16* lp = &As[buf][(c >> 2) * LDT + (c & 3) * 8];
            __builtin_amdgcn_global_load_async_to_lds_b128(
                (as1_v4i*)(as1_void*)(void*)p,
                (as3_v4i*)(as3_void*)(void*)lp, 0, 0);
        }
#pragma unroll
        for (int i = 0; i < BCH; ++i) {
            const int c = t + i * 256;
            const _Float16* p = Bg + (size_t)(c >> 2) * ldb + (c & 3) * 8 + k0;
            _Float16* lp = &Bs[buf][(c >> 2) * LDT + (c & 3) * 8];
            __builtin_amdgcn_global_load_async_to_lds_b128(
                (as1_v4i*)(as1_void*)(void*)p,
                (as3_v4i*)(as3_void*)(void*)lp, 0, 0);
        }
    };
#else
    uint4 areg[ACH], breg[BCH];
    auto gload = [&](int k0) {
#pragma unroll
        for (int i = 0; i < ACH; ++i) {
            const int c = t + i * 256;
            const _Float16* p = Ag + (size_t)(c >> 2) * lda + (c & 3) * 8 + k0;
            areg[i] = *(const uint4*)p;
            if (i == 0) __builtin_prefetch(p + 32, 0, 1);
        }
#pragma unroll
        for (int i = 0; i < BCH; ++i) {
            const int c = t + i * 256;
            const _Float16* p = Bg + (size_t)(c >> 2) * ldb + (c & 3) * 8 + k0;
            breg[i] = *(const uint4*)p;
            if (i == 0) __builtin_prefetch(p + 32, 0, 1);
        }
    };
    auto sstore = [&](int buf) {
#pragma unroll
        for (int i = 0; i < ACH; ++i) {
            const int c = t + i * 256;
            *(uint4*)&As[buf][(c >> 2) * LDT + (c & 3) * 8] = areg[i];
        }
#pragma unroll
        for (int i = 0; i < BCH; ++i) {
            const int c = t + i * 256;
            *(uint4*)&Bs[buf][(c >> 2) * LDT + (c & 3) * 8] = breg[i];
        }
    };
#endif

    v8f acc[WMT][WNT];
#pragma unroll
    for (int i = 0; i < WMT; ++i)
#pragma unroll
        for (int j = 0; j < WNT; ++j) acc[i][j] = {};

    const int nsteps = K / 32;

#if GEMM_ASYNC
    aissue(0, 0);
#else
    gload(0);
    sstore(0);
#endif

    for (int s = 0; s < nsteps; ++s) {
        const int cur = s & 1;
#if GEMM_ASYNC
        __builtin_amdgcn_s_wait_asynccnt(0);   // own slab-s DMAs landed in LDS
        __syncthreads();                       // everyone's landed; nxt buf free
        if (s + 1 < nsteps) aissue((s + 1) * 32, cur ^ 1);  // overlaps WMMAs
#else
        __syncthreads();                       // buf[s&1] ready for all waves
        if (s + 1 < nsteps) gload((s + 1) * 32);
#endif

        // Load ALL fragments first, then burst the WMMAs back-to-back.
        Frag16 a[WMT], b[WNT];
#pragma unroll
        for (int i = 0; i < WMT; ++i) {
            // A fragment: row ((wm*WMT+i)*16+lr); elems 0..7 -> K=8*hf+e,
            //             elems 8..15 -> K=16+8*hf+(e-8)
            const _Float16* ap = &As[cur][((wm * WMT + i) * 16 + lr) * LDT];
            a[i].h[0] = *(const v8h*)(ap + 8 * hf);
            a[i].h[1] = *(const v8h*)(ap + 16 + 8 * hf);
        }
#pragma unroll
        for (int j = 0; j < WNT; ++j) {
            // B fragment: W row ((wn*WNT+j)*16+lr); elems 0..7 -> K=8*hf+e,
            //             elems 8..15 -> K=16+8*hf+(e-8)
            const _Float16* bp = &Bs[cur][((wn * WNT + j) * 16 + lr) * LDT];
            b[j].h[0] = *(const v8h*)(bp + 8 * hf);
            b[j].h[1] = *(const v8h*)(bp + 16 + 8 * hf);
        }
#pragma unroll
        for (int i = 0; i < WMT; ++i)
#pragma unroll
            for (int j = 0; j < WNT; ++j)
                acc[i][j] = __builtin_amdgcn_wmma_f32_16x16x32_f16(
                                false, a[i].v, false, b[j].v, (short)0,
                                acc[i][j], false, false);

#if !GEMM_ASYNC
        if (s + 1 < nsteps) sstore(cur ^ 1);   // other buffer; next barrier protects it
#endif
    }

    // C/D layout: elem r -> C[m = 8*hf + r][n = lane%16]
#pragma unroll
    for (int i = 0; i < WMT; ++i) {
        const int cm = rowBlock + (wm * WMT + i) * 16 + 8 * hf;
#pragma unroll
        for (int j = 0; j < WNT; ++j) {
            const int cn = colBlock + (wn * WNT + j) * 16 + lr;
            const float bvv = HAS_BIAS ? bias[cn] : 0.0f;
#pragma unroll
            for (int r = 0; r < 8; ++r) {
                float v = acc[i][j][r] + bvv;
                if (HAS_RES) v += Res[(size_t)(cm + r) * ldres + cn];
                C[(size_t)(cm + r) * ldc + cn] = v;
            }
        }
    }
}

// ---------------------------------------------------------------------------
// RMSNorm: one block per row of D elements; writes f16 (and optional f32)
// ---------------------------------------------------------------------------
__global__ __launch_bounds__(256)
void rmsnorm_kern(const float* __restrict__ x, const float* __restrict__ g,
                  _Float16* __restrict__ o16, float* __restrict__ o32, int D)
{
    __shared__ float red[256];
    const size_t base = (size_t)blockIdx.x * D;
    float s = 0.f;
    for (int i = threadIdx.x; i < D; i += 256) { float v = x[base + i]; s += v * v; }
    red[threadIdx.x] = s;
    __syncthreads();
    for (int off = 128; off > 0; off >>= 1) {
        if (threadIdx.x < off) red[threadIdx.x] += red[threadIdx.x + off];
        __syncthreads();
    }
    const float rs = rsqrtf(red[0] / (float)D + 1.1920929e-07f);
    for (int i = threadIdx.x; i < D; i += 256) {
        float v = x[base + i] * rs * g[i];
        o16[base + i] = (_Float16)v;
        if (o32) o32[base + i] = v;
    }
}

// ---------------------------------------------------------------------------
// Fused RoPE + scale + head-major relayout.
//  in : q,k,v f32 [B,S,H*DH]
//  out: q16 (scaled 1/8), k16 as [B,H,S,DH]; v16t as [B,H,DH,S] (so P.V GEMM
//       can use the same W[N,K] operand layout)
// ---------------------------------------------------------------------------
__global__ void rope_qkv_kern(const float* __restrict__ qf,
                              const float* __restrict__ kf,
                              const float* __restrict__ vf,
                              _Float16* __restrict__ q16,
                              _Float16* __restrict__ k16,
                              _Float16* __restrict__ v16t, int total)
{
    int idx = blockIdx.x * 256 + threadIdx.x;
    if (idx >= total) return;
    const int d  = idx % DD;
    const int dh = d & (DHE - 1);
    const int h  = d >> 6;
    const int s  = (idx / DD) % SS;
    const int b  = idx / (DD * SS);

    const int j = dh & 31;
    const float freq  = __powf(10000.f, -(float)(2 * j) / 64.f);
    const float angle = (float)s * freq;
    float sn, cs;
    __sincosf(angle, &sn, &cs);

    const int pidx = idx + ((dh < 32) ? 32 : -32);
    const float qv = qf[idx], kv = kf[idx];
    const float qr = (dh < 32) ? -qf[pidx] : qf[pidx];
    const float kr = (dh < 32) ? -kf[pidx] : kf[pidx];

    const size_t qi = (((size_t)b * HH + h) * SS + s) * DHE + dh;
    q16[qi] = (_Float16)((qv * cs + qr * sn) * 0.125f);   // * 1/sqrt(64)
    k16[qi] = (_Float16)(kv * cs + kr * sn);
    const size_t vi = (((size_t)b * HH + h) * DHE + dh) * SS + s;
    v16t[vi] = (_Float16)vf[idx];
}

// ---------------------------------------------------------------------------
// Row softmax (row length N), f32 in -> f16 probabilities out
// ---------------------------------------------------------------------------
__global__ __launch_bounds__(256)
void softmax_kern(const float* __restrict__ x, _Float16* __restrict__ p, int N)
{
    __shared__ float red[256];
    const size_t base = (size_t)blockIdx.x * N;
    float m = -3.4e38f;
    for (int i = threadIdx.x; i < N; i += 256) m = fmaxf(m, x[base + i]);
    red[threadIdx.x] = m;
    __syncthreads();
    for (int off = 128; off > 0; off >>= 1) {
        if (threadIdx.x < off)
            red[threadIdx.x] = fmaxf(red[threadIdx.x], red[threadIdx.x + off]);
        __syncthreads();
    }
    m = red[0];
    __syncthreads();
    float s = 0.f;
    for (int i = threadIdx.x; i < N; i += 256) s += __expf(x[base + i] - m);
    red[threadIdx.x] = s;
    __syncthreads();
    for (int off = 128; off > 0; off >>= 1) {
        if (threadIdx.x < off) red[threadIdx.x] += red[threadIdx.x + off];
        __syncthreads();
    }
    const float inv = 1.f / red[0];
    for (int i = threadIdx.x; i < N; i += 256)
        p[base + i] = (_Float16)(__expf(x[base + i] - m) * inv);
}

__global__ void cvt_f32_f16(const float* __restrict__ s,
                            _Float16* __restrict__ d, int n)
{
    int i = blockIdx.x * 256 + threadIdx.x;
    if (i < n) d[i] = (_Float16)s[i];
}

// g = a * silu(b) -> f16
__global__ void silu_mul_kern(const float* __restrict__ a,
                              const float* __restrict__ b,
                              _Float16* __restrict__ g, int n)
{
    int i = blockIdx.x * 256 + threadIdx.x;
    if (i < n) {
        float bv = b[i];
        g[i] = (_Float16)(a[i] * bv / (1.f + __expf(-bv)));
    }
}

// ---------------------------------------------------------------------------
template<int WM, int WN, int WMT, int WNT>
static inline void gemm_launch_t(const _Float16* A, int lda, const _Float16* Wm, int ldb,
                                 float* C, int ldc, const float* bias,
                                 const float* Res, int ldres,
                                 int M, int N, int K, hipStream_t stream)
{
    dim3 g(N / (WN * 16 * WNT), M / (WM * 16 * WMT));
    if (bias && Res)
        gemm_wmma<WM, WN, WMT, WNT, true, true><<<g, 256, 0, stream>>>(A, lda, Wm, ldb, C, ldc, bias, Res, ldres, K);
    else if (bias)
        gemm_wmma<WM, WN, WMT, WNT, true, false><<<g, 256, 0, stream>>>(A, lda, Wm, ldb, C, ldc, bias, nullptr, 0, K);
    else if (Res)
        gemm_wmma<WM, WN, WMT, WNT, false, true><<<g, 256, 0, stream>>>(A, lda, Wm, ldb, C, ldc, nullptr, Res, ldres, K);
    else
        gemm_wmma<WM, WN, WMT, WNT, false, false><<<g, 256, 0, stream>>>(A, lda, Wm, ldb, C, ldc, nullptr, nullptr, 0, K);
}

extern "C" void kernel_launch(void* const* d_in, const int* in_sizes, int n_in,
                              void* d_out, int out_size, void* d_ws, size_t ws_size,
                              hipStream_t stream)
{
    (void)in_sizes; (void)n_in; (void)out_size; (void)ws_size;

    const float* src   = (const float*)d_in[0];
    const float* wq    = (const float*)d_in[1];
    const float* bq    = (const float*)d_in[2];
    const float* wk    = (const float*)d_in[3];
    const float* bk    = (const float*)d_in[4];
    const float* wv    = (const float*)d_in[5];
    const float* bv    = (const float*)d_in[6];
    const float* wo    = (const float*)d_in[7];
    const float* bo    = (const float*)d_in[8];
    const float* wfc1  = (const float*)d_in[9];
    const float* bfc1  = (const float*)d_in[10];
    const float* win   = (const float*)d_in[11];
    const float* wgate = (const float*)d_in[12];
    const float* wfc2  = (const float*)d_in[13];
    const float* bfc2  = (const float*)d_in[14];
    const float* gattn = (const float*)d_in[15];
    const float* gffn  = (const float*)d_in[16];
    float* out = (float*)d_out;

    const size_t MSD = (size_t)BB * SS * DD;   // 3,145,728
    const size_t MSF = (size_t)BB * SS * FF;   // 12,582,912

    char* bump = (char*)d_ws;
    auto alloc = [&](size_t bytes) -> char* {
        char* r = bump; bump += (bytes + 255) & ~(size_t)255; return r;
    };

    _Float16* x16     = (_Float16*)alloc(MSD * 2);
    _Float16* wq16    = (_Float16*)alloc((size_t)DD * DD * 2);
    _Float16* wk16    = (_Float16*)alloc((size_t)DD * DD * 2);
    _Float16* wv16    = (_Float16*)alloc((size_t)DD * DD * 2);
    _Float16* wo16    = (_Float16*)alloc((size_t)DD * DD * 2);
    _Float16* wfc1_16 = (_Float16*)alloc((size_t)FF * DD * 2);
    _Float16* win16   = (_Float16*)alloc((size_t)FF * FF * 2);
    _Float16* wgate16 = (_Float16*)alloc((size_t)FF * FF * 2);
    _Float16* wfc2_16 = (_Float16*)alloc((size_t)DD * FF * 2);
    float*    qf      = (float*)alloc(MSD * 4);
    float*    kf      = (float*)alloc(MSD * 4);
    float*    vf      = (float*)alloc(MSD * 4);
    _Float16* q16     = (_Float16*)alloc(MSD * 2);
    _Float16* k16     = (_Float16*)alloc(MSD * 2);
    _Float16* v16t    = (_Float16*)alloc(MSD * 2);
    float*    scoresf = (float*)alloc((size_t)SS * SS * 4);
    _Float16* p16     = (_Float16*)alloc((size_t)SS * SS * 2);
    float*    ctxf    = (float*)alloc(MSD * 4);
    _Float16* ctx16   = (_Float16*)alloc(MSD * 2);
    float*    zf      = (float*)alloc(MSD * 4);
    float*    yf      = (float*)alloc(MSD * 4);
    _Float16* y16     = (_Float16*)alloc(MSD * 2);
    float*    hf      = (float*)alloc(MSF * 4);
    _Float16* h16     = (_Float16*)alloc(MSF * 2);
    float*    af      = (float*)alloc(MSF * 4);
    float*    bfu     = (float*)alloc(MSF * 4);
    _Float16* g16     = (_Float16*)alloc(MSF * 2);

    auto cvt = [&](const float* s, _Float16* d, size_t n) {
        cvt_f32_f16<<<dim3((unsigned)((n + 255) / 256)), dim3(256), 0, stream>>>(s, d, (int)n);
    };
    // big GEMMs: 128x128 block tile (waves 4x2, 2x4 register blocking,
    // 8 WMMAs per wave per K-step)
    auto gemm = [&](const _Float16* A, int lda, const _Float16* Wm, int ldb,
                    float* C, int ldc, const float* bias,
                    const float* Res, int ldres, int M, int N, int K) {
        gemm_launch_t<4, 2, 2, 4>(A, lda, Wm, ldb, C, ldc, bias, Res, ldres, M, N, K, stream);
    };
    // context GEMM: N=64 -> 128x64 block tile (waves 4x2, 2x2 register blocking)
    auto gemm_n64 = [&](const _Float16* A, int lda, const _Float16* Wm, int ldb,
                        float* C, int ldc, int M, int K) {
        gemm_launch_t<4, 2, 2, 2>(A, lda, Wm, ldb, C, ldc, nullptr, nullptr, 0, M, 64, K, stream);
    };

    // ---- weights -> f16 ----
    cvt(wq,    wq16,    (size_t)DD * DD);
    cvt(wk,    wk16,    (size_t)DD * DD);
    cvt(wv,    wv16,    (size_t)DD * DD);
    cvt(wo,    wo16,    (size_t)DD * DD);
    cvt(wfc1,  wfc1_16, (size_t)FF * DD);
    cvt(win,   win16,   (size_t)FF * FF);
    cvt(wgate, wgate16, (size_t)FF * FF);
    cvt(wfc2,  wfc2_16, (size_t)DD * FF);

    // ---- attention ----
    rmsnorm_kern<<<BB * SS, 256, 0, stream>>>(src, gattn, x16, nullptr, DD);

    gemm(x16, DD, wq16, DD, qf, DD, bq, nullptr, 0, BB * SS, DD, DD);
    gemm(x16, DD, wk16, DD, kf, DD, bk, nullptr, 0, BB * SS, DD, DD);
    gemm(x16, DD, wv16, DD, vf, DD, bv, nullptr, 0, BB * SS, DD, DD);

    rope_qkv_kern<<<dim3((unsigned)((MSD + 255) / 256)), 256, 0, stream>>>(
        qf, kf, vf, q16, k16, v16t, (int)MSD);

    for (int bh = 0; bh < BB * HH; ++bh) {
        const _Float16* qh = q16  + (size_t)bh * SS * DHE;
        const _Float16* kh = k16  + (size_t)bh * SS * DHE;
        const _Float16* vh = v16t + (size_t)bh * DHE * SS;
        // scores = (q*scale) . k^T   [S,S], K=64
        gemm(qh, DHE, kh, DHE, scoresf, SS, nullptr, nullptr, 0, SS, SS, DHE);
        softmax_kern<<<SS, 256, 0, stream>>>(scoresf, p16, SS);
        // ctx = P . V   [S,64], K=S ; write straight into [B,S,H*DH] layout
        const int b = bh / HH, h = bh % HH;
        float* ctxh = ctxf + (size_t)b * SS * DD + (size_t)h * DHE;
        gemm_n64(p16, SS, vh, SS, ctxh, DD, SS, SS);
    }

    cvt(ctxf, ctx16, MSD);
    // z = src + ctx @ wo^T + bo
    gemm(ctx16, DD, wo16, DD, zf, DD, bo, src, DD, BB * SS, DD, DD);

    // ---- FFN ----
    rmsnorm_kern<<<BB * SS, 256, 0, stream>>>(zf, gffn, y16, yf, DD);

    gemm(y16, DD, wfc1_16, DD, hf, FF, bfc1, nullptr, 0, BB * SS, FF, DD);
    cvt(hf, h16, MSF);

    gemm(h16, FF, win16,   FF, af,  FF, nullptr, nullptr, 0, BB * SS, FF, FF);
    gemm(h16, FF, wgate16, FF, bfu, FF, nullptr, nullptr, 0, BB * SS, FF, FF);

    silu_mul_kern<<<dim3((unsigned)((MSF + 255) / 256)), 256, 0, stream>>>(
        af, bfu, g16, (int)MSF);

    // out = y + g @ wfc2^T + bfc2
    gemm(g16, FF, wfc2_16, FF, out, DD, bfc2, yf, DD, BB * SS, DD, FF);
}
